// MoE_72713796321590
// MI455X (gfx1250) — compile-verified, matrix operand order checked by reading, not run on hardware
//
#include <hip/hip_runtime.h>

// ---------------------------------------------------------------------------
// MoE forward for MI455X (gfx1250): top-2 routed FFN with bf16 WMMA compute.
//   E=8 experts, H=1024, F=4096, T=8192 tokens, K=2.
//
// Pipeline per launch (all deterministic, graph-capture safe):
//   1) transpose+convert Wg/Wu/Wd f32 -> bf16, N-major (so WMMA B fragments
//      become two 16-byte contiguous loads per lane)
//   2) gating (wave32 shuffle reduction, top-2 + 2-way softmax)
//   3) routing into fixed per-expert slot regions (int atomics only)
//   4) fused expert FFN: x@Wg, x@Wu -> SiLU*up -> @Wd, all via
//      v_wmma_f32_16x16x32_bf16, rank-keyed scatter (no float atomics)
//   5) combine the two weighted expert outputs per token
// ---------------------------------------------------------------------------

#define NEXP  8
#define HDIM  1024
#define FDIM  4096
#define NTOK  8192
#define CHUNK 2048   // F processed in 2 chunks so bf16 h-tile fits in LDS

typedef __attribute__((ext_vector_type(16))) __bf16 v16bf;
typedef __attribute__((ext_vector_type(8)))  __bf16 v8bf;
typedef __attribute__((ext_vector_type(8)))  float  v8f;

// ---------------------------------------------------------------------------
// Fragment load: both A (16xK) and B (KxN, stored N-major) layouts reduce to
// "row pointer + two contiguous 8-element bf16 runs at k0+half*8 and
// k0+16+half*8" (ISA 7.12.2). Each run is one 16-byte load.
// ---------------------------------------------------------------------------
static __device__ inline v16bf load_frag_run(const __bf16* row, int k0, int half) {
  const __bf16* p = row + k0 + half * 8;
  const v8bf lo = *(const v8bf*)(p);        // k = k0+half*8   .. +7
  const v8bf hi = *(const v8bf*)(p + 16);   // k = k0+16+half*8.. +7
  v16bf r;
#pragma unroll
  for (int i = 0; i < 8; ++i) { r[i] = lo[i]; r[8 + i] = hi[i]; }
  return r;
}

static __device__ inline v16bf lds_a_frag(const __bf16* base, int ld, int k0, int lane) {
  return load_frag_run(base + (size_t)(lane & 15) * ld, k0, lane >> 4);
}

static __device__ inline v16bf ldg_b_frag(const __bf16* Bt, int ldk, int k0,
                                          int n0, int lane) {
  return load_frag_run(Bt + (size_t)(n0 + (lane & 15)) * ldk, k0, lane >> 4);
}

// ---------------------------------------------------------------------------
// Kernel 0: tiled transpose + f32->bf16 convert.  src[e][r][c] -> dst[e][c][r]
// grid = (C/32, R/32, E), 256 threads, LDS 32x33 to kill bank conflicts.
// ---------------------------------------------------------------------------
__global__ __launch_bounds__(256) void transpose_cvt_kernel(
    const float* __restrict__ src, __bf16* __restrict__ dst, int R, int C) {
  __shared__ __bf16 tile[32][33];
  const size_t base = (size_t)blockIdx.z * R * C;
  const int c0 = blockIdx.x * 32, r0 = blockIdx.y * 32;
  const int tx = threadIdx.x & 31, ty = threadIdx.x >> 5;   // 8 rows/pass
#pragma unroll
  for (int i = 0; i < 32; i += 8) {
    const int r = r0 + ty + i;
    tile[ty + i][tx] = (__bf16)src[base + (size_t)r * C + c0 + tx];
  }
  __syncthreads();
#pragma unroll
  for (int i = 0; i < 32; i += 8) {
    const int c = c0 + ty + i;
    dst[base + (size_t)c * R + r0 + tx] = tile[tx][ty + i];
  }
}

// ---------------------------------------------------------------------------
// Kernel 1: gating. One wave per token (wave32). Lanes split H, cross-lane
// reduce with __shfl_xor, lane 0 does top-2 + softmax-over-top-2 weights.
// ---------------------------------------------------------------------------
__global__ __launch_bounds__(256) void gate_kernel(
    const float* __restrict__ x, const float* __restrict__ gate_w,
    const float* __restrict__ gate_b, float* __restrict__ logits_out,
    int* __restrict__ route_idx, float* __restrict__ route_w) {
  const int lane = threadIdx.x & 31;
  const int t    = blockIdx.x * 8 + (threadIdx.x >> 5);

  float acc[NEXP];
#pragma unroll
  for (int e = 0; e < NEXP; ++e) acc[e] = 0.f;

  const float* xr = x + (size_t)t * HDIM;
  for (int h = lane; h < HDIM; h += 32) {
    const float xv = xr[h];
    const float* g = gate_w + h * NEXP;
#pragma unroll
    for (int e = 0; e < NEXP; ++e) acc[e] += xv * g[e];
  }
#pragma unroll
  for (int e = 0; e < NEXP; ++e) {
#pragma unroll
    for (int off = 16; off > 0; off >>= 1)
      acc[e] += __shfl_xor(acc[e], off, 32);
  }

  if (lane == 0) {
    float v[NEXP];
#pragma unroll
    for (int e = 0; e < NEXP; ++e) {
      v[e] = acc[e] + gate_b[e];
      logits_out[(size_t)t * NEXP + e] = v[e];
    }
    // top-2 (first occurrence wins ties, matches lax.top_k)
    float v0 = -3.4e38f; int i0 = 0;
#pragma unroll
    for (int e = 0; e < NEXP; ++e) if (v[e] > v0) { v0 = v[e]; i0 = e; }
    float v1 = -3.4e38f; int i1 = 0;
#pragma unroll
    for (int e = 0; e < NEXP; ++e) if (e != i0 && v[e] > v1) { v1 = v[e]; i1 = e; }
    // softmax over {v0, v1} (all other logits are -inf in the reference)
    const float e1 = __expf(v1 - v0);
    const float inv = __builtin_amdgcn_rcpf(1.f + e1);
    route_idx[2 * t]     = i0;  route_w[2 * t]     = inv;
    route_idx[2 * t + 1] = i1;  route_w[2 * t + 1] = e1 * inv;
  }
}

// ---------------------------------------------------------------------------
// Kernel 2: routing. Fixed per-expert slot regions (stride NTOK), integer
// atomics only; slot ORDER is nondeterministic but final output is keyed by
// (token, rank), so results are bit-deterministic.
// ---------------------------------------------------------------------------
__global__ void route_kernel(const int* __restrict__ route_idx,
                             const float* __restrict__ route_w,
                             int* __restrict__ counts,
                             int* __restrict__ pair_tok,
                             float* __restrict__ pair_w) {
  const int t = blockIdx.x * blockDim.x + threadIdx.x;
  if (t >= NTOK) return;
#pragma unroll
  for (int k = 0; k < 2; ++k) {
    const int e   = route_idx[2 * t + k];
    const float w = route_w[2 * t + k];
    const int i = atomicAdd(&counts[e], 1);
    pair_tok[e * NTOK + i] = (t << 1) | k;   // token id + rank in LSB
    pair_w[e * NTOK + i]   = w;
  }
}

// ---------------------------------------------------------------------------
// Kernel 3: fused expert FFN via bf16 WMMA.
// Block = (16-token tile of one expert), 8 waves / 256 threads.
//   s_x: [16][H]     bf16 (32 KB)  - A matrix for gate/up GEMMs
//   s_h: [16][CHUNK] bf16 (64 KB)  - silu(xWg)*(xWu) chunk, A matrix for down
// Weights come from the N-major bf16 copies: WgT/WuT [e][F][H], WdT [e][H][F].
// ---------------------------------------------------------------------------
__global__ __launch_bounds__(256) void moe_ffn_kernel(
    const float* __restrict__ x, const __bf16* __restrict__ WgT,
    const __bf16* __restrict__ WuT, const __bf16* __restrict__ WdT,
    const int* __restrict__ pair_tok, const float* __restrict__ pair_w,
    const int* __restrict__ counts, float* __restrict__ y_pairs) {
  const int e   = blockIdx.y;
  const int cnt = counts[e];
  const int tile0 = blockIdx.x * 16;
  if (tile0 >= cnt) return;                    // oversized grid: early exit
  const int nvalid = min(16, cnt - tile0);

  extern __shared__ char smem[];
  __bf16* s_x = (__bf16*)smem;                       // 16 x HDIM
  __bf16* s_h = (__bf16*)(smem + 16 * HDIM * 2);     // 16 x CHUNK

  const int tid  = threadIdx.x;
  const int lane = tid & 31;
  const int wv   = tid >> 5;

  const int* ptok = pair_tok + e * NTOK + tile0;

  // Stage x rows for this tile into LDS as bf16 (invalid rows dup row 0).
  for (int idx = tid; idx < 16 * HDIM; idx += 256) {
    const int r = idx >> 10, c = idx & (HDIM - 1);
    const int rr  = (r < nvalid) ? r : 0;
    const int tok = ptok[rr] >> 1;
    s_x[idx] = (__bf16)x[(size_t)tok * HDIM + c];
  }
  __syncthreads();

  const __bf16* WgT_e = WgT + (size_t)e * HDIM * FDIM;   // [F][H] n-major
  const __bf16* WuT_e = WuT + (size_t)e * HDIM * FDIM;   // [F][H] n-major
  const __bf16* WdT_e = WdT + (size_t)e * FDIM * HDIM;   // [H][F] n-major

  const v8f vzero = {};
  v8f accY[8];
#pragma unroll
  for (int i = 0; i < 8; ++i) accY[i] = vzero;

  for (int ch = 0; ch < FDIM / CHUNK; ++ch) {
    const int cbase = ch * CHUNK;

    // ---- GEMM1: h = silu(x @ Wg) * (x @ Wu), this wave's 256 cols ----
    for (int ct = 0; ct < 16; ++ct) {
      const int nloc = wv * 256 + ct * 16;   // column within chunk
      const int ng   = cbase + nloc;         // column within F
      v8f ag = vzero, au = vzero;
      for (int k0 = 0; k0 < HDIM; k0 += 32) {
        const v16bf a  = lds_a_frag(s_x, HDIM, k0, lane);
        const v16bf bg = ldg_b_frag(WgT_e, HDIM, k0, ng, lane);
        const v16bf bu = ldg_b_frag(WuT_e, HDIM, k0, ng, lane);
        ag = __builtin_amdgcn_wmma_f32_16x16x32_bf16(false, a, false, bg,
                                                     (short)0, ag, false, false);
        au = __builtin_amdgcn_wmma_f32_16x16x32_bf16(false, a, false, bu,
                                                     (short)0, au, false, false);
      }
      // fused SiLU * up via fast v_rcp_f32; C/D layout: row = r + 8*half
      const int row_hi = (lane >> 4) * 8;
      const int col    = nloc + (lane & 15);
#pragma unroll
      for (int r = 0; r < 8; ++r) {
        const float g = ag[r], u = au[r];
        const float hval = g * u * __builtin_amdgcn_rcpf(1.f + __expf(-g));
        s_h[(row_hi + r) * CHUNK + col] = (__bf16)hval;
      }
    }
    __syncthreads();

    // ---- GEMM2 partial: y += h_chunk @ Wd[cbase:cbase+CHUNK, :] ----
#pragma unroll
    for (int ct = 0; ct < 8; ++ct) {
      const int n0 = wv * 128 + ct * 16;
      v8f acc = accY[ct];
      for (int k0 = 0; k0 < CHUNK; k0 += 32) {
        const v16bf a = lds_a_frag(s_h, CHUNK, k0, lane);
        const v16bf b = ldg_b_frag(WdT_e, FDIM, cbase + k0, n0, lane);
        acc = __builtin_amdgcn_wmma_f32_16x16x32_bf16(false, a, false, b,
                                                      (short)0, acc, false, false);
      }
      accY[ct] = acc;
    }
    __syncthreads();   // s_h reuse across chunks
  }

  // Weighted scatter to y_pairs[token*2 + rank][:] (deterministic, no atomics)
  const float* pw = pair_w + e * NTOK + tile0;
  const int row_hi = (lane >> 4) * 8;
  const int col_lo = lane & 15;
#pragma unroll
  for (int ct = 0; ct < 8; ++ct) {
    const int col = wv * 128 + ct * 16 + col_lo;
#pragma unroll
    for (int r = 0; r < 8; ++r) {
      const int row = row_hi + r;
      if (row < nvalid) {
        const int tr    = ptok[row];   // (token<<1)|rank
        const float wgt = pw[row];
        y_pairs[(size_t)tr * HDIM + col] = wgt * accY[ct][r];
      }
    }
  }
}

// ---------------------------------------------------------------------------
// Kernel 4: combine the two weighted expert outputs per token.
// ---------------------------------------------------------------------------
__global__ void combine_kernel(const float* __restrict__ y_pairs,
                               float* __restrict__ out) {
  const int idx = blockIdx.x * 256 + threadIdx.x;     // t*HDIM + h
  if (idx >= NTOK * HDIM) return;
  const int t = idx >> 10;
  const int h = idx & (HDIM - 1);
  out[idx] = y_pairs[(size_t)(2 * t) * HDIM + h] +
             y_pairs[(size_t)(2 * t + 1) * HDIM + h];
}

// ---------------------------------------------------------------------------
// Launch: all on `stream`, graph-capture safe (hipMemsetAsync only).
// Workspace layout (bytes):
//   [0)      route_idx  int[2T]        = 64 KB
//   [64K)    route_w    f32[2T]        = 64 KB
//   [128K)   counts     int[E]
//   [129K)   pair_tok   int[E*T]       = 256 KB
//   [385K)   pair_w     f32[E*T]       = 256 KB
//   [1M)     y_pairs    f32[2T*H]      = 64 MB
//   [65M)    WgT        bf16[E*F*H]    = 64 MB
//   [129M)   WuT        bf16[E*F*H]    = 64 MB
//   [193M)   WdT        bf16[E*H*F]    = 64 MB   (total 257 MB)
// ---------------------------------------------------------------------------
extern "C" void kernel_launch(void* const* d_in, const int* in_sizes, int n_in,
                              void* d_out, int out_size, void* d_ws, size_t ws_size,
                              hipStream_t stream) {
  const float* x      = (const float*)d_in[0];
  const float* gate_w = (const float*)d_in[1];
  const float* gate_b = (const float*)d_in[2];
  const float* Wg     = (const float*)d_in[3];
  const float* Wu     = (const float*)d_in[4];
  const float* Wd     = (const float*)d_in[5];

  float* out_acc    = (float*)d_out;                       // [T, H]
  float* out_logits = out_acc + (size_t)NTOK * HDIM;       // [T, E]

  char* ws = (char*)d_ws;
  int*    route_idx = (int*)(ws);
  float*  route_w   = (float*)(ws + 65536);
  int*    counts    = (int*)(ws + 131072);
  int*    pair_tok  = (int*)(ws + 132096);
  float*  pair_w    = (float*)(ws + 394240);
  float*  y_pairs   = (float*)(ws + (1ull << 20));
  const size_t MB   = 1ull << 20;
  __bf16* WgT       = (__bf16*)(ws + 65 * MB);
  __bf16* WuT       = (__bf16*)(ws + 129 * MB);
  __bf16* WdT       = (__bf16*)(ws + 193 * MB);

  // Weight transpose+convert: Wg/Wu [e][H][F] -> [e][F][H]; Wd [e][F][H] -> [e][H][F]
  transpose_cvt_kernel<<<dim3(FDIM / 32, HDIM / 32, NEXP), 256, 0, stream>>>(
      Wg, WgT, HDIM, FDIM);
  transpose_cvt_kernel<<<dim3(FDIM / 32, HDIM / 32, NEXP), 256, 0, stream>>>(
      Wu, WuT, HDIM, FDIM);
  transpose_cvt_kernel<<<dim3(HDIM / 32, FDIM / 32, NEXP), 256, 0, stream>>>(
      Wd, WdT, FDIM, HDIM);

  hipMemsetAsync(counts, 0, NEXP * sizeof(int), stream);

  gate_kernel<<<NTOK / 8, 256, 0, stream>>>(x, gate_w, gate_b, out_logits,
                                            route_idx, route_w);
  route_kernel<<<NTOK / 256, 256, 0, stream>>>(route_idx, route_w, counts,
                                               pair_tok, pair_w);

  const size_t smem_bytes = (size_t)(16 * HDIM + 16 * CHUNK) * 2;  // 96 KB
  moe_ffn_kernel<<<dim3(NTOK / 16, NEXP), 256, smem_bytes, stream>>>(
      x, WgT, WuT, WdT, pair_tok, pair_w, counts, y_pairs);

  combine_kernel<<<(NTOK * HDIM) / 256, 256, 0, stream>>>(y_pairs, out_acc);
}
